// cfmf_7816840479105
// MI455X (gfx1250) — matrix operand order repaired and checked by reading, not compiled, not run on hardware
//
#include <hip/hip_runtime.h>
#include <math.h>

// ---------------- types for WMMA (gfx1250, wave32) ----------------
typedef __attribute__((ext_vector_type(16))) __bf16 v16bf;
typedef __attribute__((ext_vector_type(8)))  float  v8f;
typedef __attribute__((ext_vector_type(4)))  unsigned u32x4;

union ABU { u32x4 q[2]; v16bf v; };

__device__ __forceinline__ unsigned short f2bf(float f) {
  unsigned u = __builtin_bit_cast(unsigned, f);
  unsigned r = (u + 0x7FFFu + ((u >> 16) & 1u)) >> 16;
  return (unsigned short)r;
}
__device__ __forceinline__ float bf2f(unsigned short h) {
  return __builtin_bit_cast(float, ((unsigned)h) << 16);
}

// A fragment: 16x32 bf16 tile, source row-major [M][K], row stride ld (halves).
// lane<16: row=lane,    K halves {0..7} then {16..23}
// lane>=16: row=lane-16, K halves {8..15} then {24..31}
__device__ __forceinline__ v16bf load_a16(const unsigned short* row0, int ld, int lane) {
  int r = lane & 15;
  const unsigned short* p = row0 + (size_t)r * ld + ((lane & 16) ? 8 : 0);
  ABU u;
  u.q[0] = *(const u32x4*)(p);
  u.q[1] = *(const u32x4*)(p + 16);
  return u.v;
}
// B fragment: 32x16 bf16 tile, source row-major [K][N], row stride ld (halves).
// col n = lane%16; lane<16: K {0..7} then {16..23}; lane>=16: K {8..15} then {24..31}.
__device__ __forceinline__ v16bf load_b16(const unsigned short* kn0, int ld, int lane) {
  int n = lane & 15;
  int h8 = (lane & 16) ? 8 : 0;
  union { unsigned short s[16]; v16bf v; } u;
  for (int s = 0; s < 8; ++s) {
    u.s[s] = kn0[(size_t)(h8 + s) * ld + n];
    u.s[8 + s] = kn0[(size_t)(16 + h8 + s) * ld + n];
  }
  return u.v;
}
__device__ __forceinline__ v8f wmma_bf(v16bf a, v16bf b, v8f c) {
  return __builtin_amdgcn_wmma_f32_16x16x32_bf16(false, a, false, b, (short)0, c, false, false);
}

#define BATCH 16
#define CH 256
#define NTOK 4096   // 64*64

// ---------------- 1) channel pooling of diff / sum ----------------
__global__ void k_pool(const float* __restrict__ x0, const float* __restrict__ x1,
                       const float* __restrict__ x2, const float* __restrict__ x3,
                       const float* __restrict__ x4, float* __restrict__ pd,
                       float* __restrict__ ps) {
  int bc = blockIdx.x;                 // b*256 + c
  size_t base = (size_t)bc * NTOK;
  float sd = 0.f, ss = 0.f;
  for (int i = threadIdx.x; i < NTOK; i += 256) {
    float a0 = x0[base + i], a1 = x1[base + i], a2 = x2[base + i];
    float a3 = x3[base + i], a4 = x4[base + i];
    float d = fabsf(fabsf(fabsf(fabsf(a0 - a1) - a2) - a3) - a4);
    sd += d;
    ss += a0 + a1 + a2 + a3 + a4;
  }
  __shared__ float rd[256], rs[256];
  rd[threadIdx.x] = sd; rs[threadIdx.x] = ss;
  __syncthreads();
  for (int s = 128; s > 0; s >>= 1) {
    if (threadIdx.x < (unsigned)s) {
      rd[threadIdx.x] += rd[threadIdx.x + s];
      rs[threadIdx.x] += rs[threadIdx.x + s];
    }
    __syncthreads();
  }
  if (threadIdx.x == 0) {
    pd[bc] = rd[0] * (1.f / NTOK);
    ps[bc] = rs[0] * (1.f / NTOK);
  }
}

// ---------------- 2) SE MLPs + softmax over 5 -> wsum[b][s][c] = wd+ws+1 ----
__global__ void k_se(const float* __restrict__ pd, const float* __restrict__ ps,
                     const float* __restrict__ sw1, const float* __restrict__ sw2,
                     const float* __restrict__ cw1, const float* __restrict__ cw2,
                     float* __restrict__ wsum) {
  int b = blockIdx.x;
  __shared__ float Pd[256], Ps[256], Hd[320], Hs[320];
  int t = threadIdx.x;
  Pd[t] = pd[b * 256 + t];
  Ps[t] = ps[b * 256 + t];
  __syncthreads();
  for (int o = t; o < 320; o += 256) {
    float ad = 0.f, as = 0.f;
    for (int c = 0; c < 256; ++c) {
      ad += sw1[o * 256 + c] * Pd[c];
      as += cw1[o * 256 + c] * Ps[c];
    }
    Hd[o] = fmaxf(ad, 0.f);
    Hs[o] = fmaxf(as, 0.f);
  }
  __syncthreads();
  int c = t;
  float od[5], oc_[5];
  for (int s = 0; s < 5; ++s) {
    int row = s * 256 + c;
    float ad = 0.f, as = 0.f;
    for (int h = 0; h < 320; ++h) {
      ad += sw2[row * 320 + h] * Hd[h];
      as += cw2[row * 320 + h] * Hs[h];
    }
    od[s] = fmaxf(ad, 0.f);
    oc_[s] = fmaxf(as, 0.f);
  }
  float md = od[0], mc = oc_[0];
  for (int s = 1; s < 5; ++s) { md = fmaxf(md, od[s]); mc = fmaxf(mc, oc_[s]); }
  float sd = 0.f, sc = 0.f;
  for (int s = 0; s < 5; ++s) {
    od[s] = expf(od[s] - md); oc_[s] = expf(oc_[s] - mc);
    sd += od[s]; sc += oc_[s];
  }
  for (int s = 0; s < 5; ++s)
    wsum[(b * 5 + s) * 256 + c] = od[s] / sd + oc_[s] / sc + 1.f;
}

// ---------------- 3) v3 = sum_s x_s * wsum  (token-major bf16) -------------
__global__ void k_v3(const float* __restrict__ x0, const float* __restrict__ x1,
                     const float* __restrict__ x2, const float* __restrict__ x3,
                     const float* __restrict__ x4, const float* __restrict__ wsum,
                     unsigned short* __restrict__ v3t) {
  const size_t total = (size_t)BATCH * CH * NTOK;
  for (size_t idx = (size_t)blockIdx.x * blockDim.x + threadIdx.x; idx < total;
       idx += (size_t)gridDim.x * blockDim.x) {
    int n = (int)(idx & (NTOK - 1));
    int c = (int)((idx >> 12) & 255);
    int b = (int)(idx >> 20);
    const float* w = wsum + (size_t)b * 5 * 256;
    float v = x0[idx] * w[c] + x1[idx] * w[256 + c] + x2[idx] * w[512 + c] +
              x3[idx] * w[768 + c] + x4[idx] * w[1024 + c];
    v3t[((size_t)(b * NTOK + n)) * CH + c] = f2bf(v);
  }
}

// ---------------- weight prep -------------------------------------
// pre_w [256][256][3][3] f32 -> [pos][c][oc] bf16
__global__ void k_prep_convw(const float* __restrict__ w, unsigned short* __restrict__ w3t) {
  for (int idx = blockIdx.x * blockDim.x + threadIdx.x; idx < 256 * 256 * 9;
       idx += gridDim.x * blockDim.x) {
    int pos = idx % 9;
    int c = (idx / 9) % 256;
    int oc = idx / (9 * 256);
    w3t[(pos * 256 + c) * 256 + oc] = f2bf(w[idx]);
  }
}
// [O][I] f32 -> [I][O] bf16
__global__ void k_transpose_bf(const float* __restrict__ w, unsigned short* __restrict__ wt,
                               int O, int I) {
  for (int idx = blockIdx.x * blockDim.x + threadIdx.x; idx < O * I;
       idx += gridDim.x * blockDim.x) {
    int i = idx % I;
    int o = idx / I;
    wt[i * O + o] = f2bf(w[idx]);
  }
}

// ---------------- 4) 3x3 conv 256->256 + bias + ReLU (WMMA) --------------
// implicit GEMM, M = pixels, N = out-ch, K = 9*256. Each wave: 2 oc-tiles
// sharing one A fragment. Boundary handled branchlessly (clamp + cndmask).
__global__ void k_conv3(const unsigned short* __restrict__ v3t,
                        const unsigned short* __restrict__ w3t,
                        const float* __restrict__ bias,
                        unsigned short* __restrict__ pre) {
  int lane = threadIdx.x & 31;
  int wv = threadIdx.x >> 5;
  int bx = blockIdx.x;
  int b = bx >> 8;
  int r = bx & 255;
  int y = r >> 2;
  int x0 = (r & 3) * 16;
  int oc0 = (blockIdx.y * 4 + wv) * 32;        // this wave: oc0 .. oc0+31
  int p = lane & 15;
  v8f acc0 = {}, acc1 = {};
  const u32x4 zz = {};
  for (int pos = 0; pos < 9; ++pos) {
    int dy = pos / 3, dx = pos % 3;
    int yy = y + dy - 1;
    if (yy < 0 || yy >= 64) continue;          // wave-uniform
    int xc = x0 + p + dx - 1;
    bool valid = (xc >= 0 && xc < 64);
    int xcl = xc < 0 ? 0 : (xc > 63 ? 63 : xc);
    const unsigned short* arow =
        v3t + ((size_t)(b * NTOK + yy * 64 + xcl)) * CH + ((lane & 16) ? 8 : 0);
    const unsigned short* bbase = w3t + (size_t)pos * 256 * 256;
    for (int kk = 0; kk < 8; ++kk) {
      int c0 = kk * 32;
      ABU ua;
      u32x4 a0 = *(const u32x4*)(arow + c0);
      u32x4 a1 = *(const u32x4*)(arow + c0 + 16);
      ua.q[0] = valid ? a0 : zz;               // branchless zero (v_cndmask)
      ua.q[1] = valid ? a1 : zz;
      v16bf b0 = load_b16(bbase + (size_t)c0 * 256 + oc0, 256, lane);
      v16bf b1 = load_b16(bbase + (size_t)c0 * 256 + oc0 + 16, 256, lane);
      acc0 = wmma_bf(ua.v, b0, acc0);
      acc1 = wmma_bf(ua.v, b1, acc1);
    }
  }
  int col = oc0 + (lane & 15);
  float bv0 = bias[col];
  float bv1 = bias[col + 16];
  int mb = (lane & 16) ? 8 : 0;
#pragma unroll
  for (int q = 0; q < 8; ++q) {
    int tok = y * 64 + x0 + q + mb;
    size_t rowoff = ((size_t)(b * NTOK + tok)) * CH;
    pre[rowoff + col]      = f2bf(fmaxf(acc0[q] + bv0, 0.f));
    pre[rowoff + col + 16] = f2bf(fmaxf(acc1[q] + bv1, 0.f));
  }
}

// ---------------- 5) tdec front: T = X@red^T ; K/V = l2norm(T@W^T+b) -------
// one wave per 16-token block; k-step outer loop, 8 accumulators reuse each A.
__global__ void k_kv(const unsigned short* __restrict__ in_tok,  // [B][4096][256]
                     const unsigned short* __restrict__ redwt,   // [256][128]
                     const unsigned short* __restrict__ wkt,     // [128][128]
                     const float* __restrict__ wkb,
                     const unsigned short* __restrict__ wvt,
                     const float* __restrict__ wvb,
                     unsigned short* __restrict__ ktok,  // [B][4096][128]
                     unsigned short* __restrict__ kT,    // [B][128][4096]
                     unsigned short* __restrict__ vtok)  // [B][4096][128]
{
  __shared__ __align__(16) unsigned short Tb[16 * 128];
  __shared__ float F[16 * 128];
  __shared__ float rn[16];
  int lane = threadIdx.x;
  int bx = blockIdx.x;
  int b = bx >> 8;
  int n0 = (bx & 255) * 16;
  const unsigned short* arow = in_tok + ((size_t)(b * NTOK + n0)) * CH;
  int col16 = lane & 15;
  int mb = (lane & 16) ? 8 : 0;
  v8f zf = {};
  // T = X(16x256) @ redwt(256x128): A loaded once per k-step, 8 WMMAs on it
  {
    v8f acc[8];
#pragma unroll
    for (int nt = 0; nt < 8; ++nt) acc[nt] = zf;
    for (int kk = 0; kk < 8; ++kk) {
      v16bf a = load_a16(arow + kk * 32, CH, lane);
#pragma unroll
      for (int nt = 0; nt < 8; ++nt) {
        v16bf bm = load_b16(redwt + (size_t)(kk * 32) * 128 + nt * 16, 128, lane);
        acc[nt] = wmma_bf(a, bm, acc[nt]);
      }
    }
#pragma unroll
    for (int nt = 0; nt < 8; ++nt) {
      int col = nt * 16 + col16;
#pragma unroll
      for (int q = 0; q < 8; ++q) Tb[(q + mb) * 128 + col] = f2bf(acc[nt][q]);
    }
  }
  // pass 0: K (writes ktok + kT), pass 1: V (writes vtok)
  for (int pass = 0; pass < 2; ++pass) {
    const unsigned short* wt = pass ? wvt : wkt;
    const float* wb = pass ? wvb : wkb;
    v8f acc[8];
#pragma unroll
    for (int nt = 0; nt < 8; ++nt) acc[nt] = zf;
    for (int kk = 0; kk < 4; ++kk) {
      v16bf a = load_a16(&Tb[kk * 32], 128, lane);
#pragma unroll
      for (int nt = 0; nt < 8; ++nt) {
        v16bf bm = load_b16(wt + (size_t)(kk * 32) * 128 + nt * 16, 128, lane);
        acc[nt] = wmma_bf(a, bm, acc[nt]);
      }
    }
#pragma unroll
    for (int nt = 0; nt < 8; ++nt) {
      int col = nt * 16 + col16;
      float bv = wb[col];
#pragma unroll
      for (int q = 0; q < 8; ++q) F[(q + mb) * 128 + col] = acc[nt][q] + bv;
    }
    if (lane < 16) {  // per-row L2 norm (F.normalize, eps 1e-12)
      float s = 0.f;
      for (int j = 0; j < 128; ++j) { float v = F[lane * 128 + j]; s += v * v; }
      rn[lane] = 1.f / fmaxf(sqrtf(s), 1e-12f);
    }
    for (int e = lane; e < 16 * 128; e += 32) {
      int row = e >> 7, j = e & 127;
      unsigned short h = f2bf(F[e] * rn[row]);
      if (pass == 0) {
        ktok[((size_t)(b * NTOK + n0 + row)) * 128 + j] = h;
        kT[((size_t)(b * 128 + j)) * NTOK + (n0 + row)] = h;
      } else {
        vtok[((size_t)(b * NTOK + n0 + row)) * 128 + j] = h;
      }
    }
  }
}

// ---------------- 6) gram: G = K^T K  (M=N=128, K=4096), 2 j-tiles/wave ----
__global__ void k_gram(const unsigned short* __restrict__ kT,
                       const unsigned short* __restrict__ ktok,
                       float* __restrict__ G) {
  int lane = threadIdx.x;
  int bx = blockIdx.x;
  int b = bx >> 5;
  int t = bx & 31;
  int i0 = (t >> 2) * 16, j0 = (t & 3) * 32;
  const unsigned short* abase = kT + ((size_t)(b * 128 + i0)) * NTOK;
  const unsigned short* bbase = ktok + ((size_t)b * NTOK) * 128 + j0;
  v8f acc0 = {}, acc1 = {};
  for (int k0 = 0; k0 < NTOK; k0 += 32) {
    v16bf a = load_a16(abase + k0, NTOK, lane);
    v16bf b0 = load_b16(bbase + (size_t)k0 * 128, 128, lane);
    v16bf b1 = load_b16(bbase + (size_t)k0 * 128 + 16, 128, lane);
    acc0 = wmma_bf(a, b0, acc0);
    acc1 = wmma_bf(a, b1, acc1);
  }
  int col = j0 + (lane & 15);
  int mb = (lane & 16) ? 8 : 0;
#pragma unroll
  for (int q = 0; q < 8; ++q) {
    size_t rowoff = ((size_t)(b * 128 + i0 + q + mb)) * 128;
    G[rowoff + col] = acc0[q];
    G[rowoff + col + 16] = acc1[q];
  }
}

// ---------------- 7) row softmax of 30*G ----------------------------------
__global__ void k_softmax(const float* __restrict__ G, float* __restrict__ A) {
  int row = blockIdx.x;  // b*128 + i
  int j = threadIdx.x;   // 128
  __shared__ float sh[128];
  float g = G[(size_t)row * 128 + j] * 30.f;
  sh[j] = g;
  __syncthreads();
  for (int s = 64; s > 0; s >>= 1) {
    if (j < s) sh[j] = fmaxf(sh[j], sh[j + s]);
    __syncthreads();
  }
  float m = sh[0];
  __syncthreads();
  float e = expf(g - m);
  sh[j] = e;
  __syncthreads();
  for (int s = 64; s > 0; s >>= 1) {
    if (j < s) sh[j] += sh[j + s];
    __syncthreads();
  }
  A[(size_t)row * 128 + j] = e / sh[0];
}

// ---------------- 8) RA^T[b][j][o] = sum_i rise[o,i]*attn[b,i,j] ----------
__global__ void k_ra(const float* __restrict__ rise, const float* __restrict__ A,
                     unsigned short* __restrict__ RAT) {
  int bj = blockIdx.x;  // b*128 + j
  int b = bj >> 7, j = bj & 127;
  int o = threadIdx.x;  // 256
  __shared__ float aj[128];
  if (o < 128) aj[o] = A[((size_t)(b * 128 + o)) * 128 + j];
  __syncthreads();
  float s = 0.f;
  for (int i = 0; i < 128; ++i) s += rise[o * 128 + i] * aj[i];
  RAT[((size_t)bj) * 256 + o] = f2bf(s);
}

// ---------------- 9) out GEMM: f = V @ RA^T + bias, 2 n-tiles/wave ---------
// mode 0 (tdec1): write f1 (bf16) and t2in = f1 + v3 (bf16, token-major)
// mode 1 (tdec2): write d_out = f3 + f1 + v3 (f32, channel-major)
__global__ void k_out(const unsigned short* __restrict__ vtok,
                      const unsigned short* __restrict__ RAT,
                      const float* __restrict__ riseb,
                      const unsigned short* __restrict__ v3t,
                      const unsigned short* __restrict__ f1_in,
                      unsigned short* __restrict__ f1_out,
                      unsigned short* __restrict__ t2in,
                      float* __restrict__ dout, int mode) {
  int lane = threadIdx.x & 31;
  int wv = threadIdx.x >> 5;
  int bx = blockIdx.x;
  int b = bx >> 8;
  int m0 = (bx & 255) * 16;
  int n0 = (blockIdx.y * 4 + wv) * 32;         // this wave: n0 .. n0+31
  const unsigned short* abase = vtok + ((size_t)(b * NTOK + m0)) * 128;
  const unsigned short* bbase = RAT + ((size_t)b * 128) * 256 + n0;
  v8f acc0 = {}, acc1 = {};
  for (int kk = 0; kk < 4; ++kk) {
    v16bf a = load_a16(abase + kk * 32, 128, lane);
    v16bf b0 = load_b16(bbase + (size_t)(kk * 32) * 256, 256, lane);
    v16bf b1 = load_b16(bbase + (size_t)(kk * 32) * 256 + 16, 256, lane);
    acc0 = wmma_bf(a, b0, acc0);
    acc1 = wmma_bf(a, b1, acc1);
  }
  int col = n0 + (lane & 15);
  float bv0 = riseb[col];
  float bv1 = riseb[col + 16];
  int mb = (lane & 16) ? 8 : 0;
#pragma unroll
  for (int q = 0; q < 8; ++q) {
    int tok = m0 + q + mb;
    size_t tm = ((size_t)(b * NTOK + tok)) * CH + col;
    float v0 = acc0[q] + bv0;
    float v1 = acc1[q] + bv1;
    if (mode == 0) {
      f1_out[tm] = f2bf(v0);
      f1_out[tm + 16] = f2bf(v1);
      t2in[tm] = f2bf(v0 + bf2f(v3t[tm]));
      t2in[tm + 16] = f2bf(v1 + bf2f(v3t[tm + 16]));
    } else {
      dout[((size_t)(b * CH + col)) * NTOK + tok] = v0 + bf2f(f1_in[tm]) + bf2f(v3t[tm]);
      dout[((size_t)(b * CH + col + 16)) * NTOK + tok] =
          v1 + bf2f(f1_in[tm + 16]) + bf2f(v3t[tm + 16]);
    }
  }
}

// ================== host launcher ==================
extern "C" void kernel_launch(void* const* d_in, const int* in_sizes, int n_in,
                              void* d_out, int out_size, void* d_ws, size_t ws_size,
                              hipStream_t stream) {
  (void)in_sizes; (void)n_in; (void)out_size; (void)ws_size;
  const float* x0 = (const float*)d_in[0];
  const float* x1 = (const float*)d_in[1];
  const float* x2 = (const float*)d_in[2];
  const float* x3 = (const float*)d_in[3];
  const float* x4 = (const float*)d_in[4];
  const float* spec_w1 = (const float*)d_in[5];
  const float* spec_w2 = (const float*)d_in[6];
  const float* comm_w1 = (const float*)d_in[7];
  const float* comm_w2 = (const float*)d_in[8];
  const float* pre_w = (const float*)d_in[9];
  const float* pre_b = (const float*)d_in[10];
  const float* wk1_w = (const float*)d_in[11];
  const float* wk1_b = (const float*)d_in[12];
  const float* wv1_w = (const float*)d_in[13];
  const float* wv1_b = (const float*)d_in[14];
  const float* red1_w = (const float*)d_in[15];
  const float* rise1_w = (const float*)d_in[16];
  const float* rise1_b = (const float*)d_in[17];
  const float* wk2_w = (const float*)d_in[18];
  const float* wk2_b = (const float*)d_in[19];
  const float* wv2_w = (const float*)d_in[20];
  const float* wv2_b = (const float*)d_in[21];
  const float* red2_w = (const float*)d_in[22];
  const float* rise2_w = (const float*)d_in[23];
  const float* rise2_b = (const float*)d_in[24];

  char* ws = (char*)d_ws;
  size_t off = 0;
  auto alloc = [&](size_t bytes) -> void* {
    void* p = ws + off;
    off += (bytes + 255) & ~(size_t)255;
    return p;
  };
  const size_t NE = (size_t)BATCH * CH * NTOK;         // 16.78M
  float* pd = (float*)alloc(16 * 256 * 4);
  float* ps = (float*)alloc(16 * 256 * 4);
  float* wsum = (float*)alloc(16 * 5 * 256 * 4);
  unsigned short* v3t = (unsigned short*)alloc(NE * 2);
  unsigned short* pre = (unsigned short*)alloc(NE * 2);   // reused as t2in
  unsigned short* f1 = (unsigned short*)alloc(NE * 2);
  unsigned short* ktok = (unsigned short*)alloc((size_t)16 * NTOK * 128 * 2);
  unsigned short* kT = (unsigned short*)alloc((size_t)16 * 128 * NTOK * 2);
  unsigned short* vtok = (unsigned short*)alloc((size_t)16 * NTOK * 128 * 2);
  float* G = (float*)alloc((size_t)16 * 128 * 128 * 4);
  float* attn = (float*)alloc((size_t)16 * 128 * 128 * 4);
  unsigned short* RAT = (unsigned short*)alloc((size_t)16 * 128 * 256 * 2);
  unsigned short* w3t = (unsigned short*)alloc((size_t)9 * 256 * 256 * 2);
  unsigned short* r1t = (unsigned short*)alloc(256 * 128 * 2);
  unsigned short* wk1t = (unsigned short*)alloc(128 * 128 * 2);
  unsigned short* wv1t = (unsigned short*)alloc(128 * 128 * 2);
  unsigned short* r2t = (unsigned short*)alloc(256 * 128 * 2);
  unsigned short* wk2t = (unsigned short*)alloc(128 * 128 * 2);
  unsigned short* wv2t = (unsigned short*)alloc(128 * 128 * 2);

  // weight prep
  k_prep_convw<<<64, 256, 0, stream>>>(pre_w, w3t);
  k_transpose_bf<<<64, 256, 0, stream>>>(red1_w, r1t, 128, 256);
  k_transpose_bf<<<64, 256, 0, stream>>>(wk1_w, wk1t, 128, 128);
  k_transpose_bf<<<64, 256, 0, stream>>>(wv1_w, wv1t, 128, 128);
  k_transpose_bf<<<64, 256, 0, stream>>>(red2_w, r2t, 128, 256);
  k_transpose_bf<<<64, 256, 0, stream>>>(wk2_w, wk2t, 128, 128);
  k_transpose_bf<<<64, 256, 0, stream>>>(wv2_w, wv2t, 128, 128);

  // SE weighting + v3
  k_pool<<<4096, 256, 0, stream>>>(x0, x1, x2, x3, x4, pd, ps);
  k_se<<<16, 256, 0, stream>>>(pd, ps, spec_w1, spec_w2, comm_w1, comm_w2, wsum);
  k_v3<<<8192, 256, 0, stream>>>(x0, x1, x2, x3, x4, wsum, v3t);

  // conv3x3 + ReLU (2 oc-tiles per wave -> grid.y = 2)
  k_conv3<<<dim3(4096, 2), 128, 0, stream>>>(v3t, w3t, pre_b, pre);

  // tdec 1
  k_kv<<<4096, 32, 0, stream>>>(pre, r1t, wk1t, wk1_b, wv1t, wv1_b, ktok, kT, vtok);
  k_gram<<<512, 32, 0, stream>>>(kT, ktok, G);
  k_softmax<<<2048, 128, 0, stream>>>(G, attn);
  k_ra<<<2048, 256, 0, stream>>>(rise1_w, attn, RAT);
  k_out<<<dim3(4096, 2), 128, 0, stream>>>(vtok, RAT, rise1_b, v3t, (const unsigned short*)nullptr,
                                           f1, pre /*t2in alias*/, (float*)nullptr, 0);

  // tdec 2 (input = f1 + v3 in `pre`)
  k_kv<<<4096, 32, 0, stream>>>(pre, r2t, wk2t, wk2_b, wv2t, wv2_b, ktok, kT, vtok);
  k_gram<<<512, 32, 0, stream>>>(kT, ktok, G);
  k_softmax<<<2048, 128, 0, stream>>>(G, attn);
  k_ra<<<2048, 256, 0, stream>>>(rise2_w, attn, RAT);
  k_out<<<dim3(4096, 2), 128, 0, stream>>>(vtok, RAT, rise2_b, v3t, f1,
                                           (unsigned short*)nullptr, (unsigned short*)nullptr,
                                           (float*)d_out, 1);
}
